// AttentionLSTM_33397665693861
// MI455X (gfx1250) — compile-verified, matrix operand order checked by reading, not run on hardware
//
#include <hip/hip_runtime.h>
#include <hip/hip_bf16.h>

// ---------------------------------------------------------------------------
// AttentionLSTM for MI455X (gfx1250, wave32, WMMA)
// Shapes: N=256, T=64, D=1024, H=1024, K = D+H+H = 3072, 4H = 4096
// ---------------------------------------------------------------------------

#define N_  256
#define T_  64
#define D_  1024
#define H_  1024
#define K_  3072   // D + H + H
#define FH_ 4096   // 4*H

typedef __attribute__((ext_vector_type(16))) __bf16 v16bf;
typedef __attribute__((ext_vector_type(8)))  __bf16 v8bf;
typedef __attribute__((ext_vector_type(8)))  float  v8f;

static __device__ inline __bf16 f2bf(float f) {
    union { float f; unsigned u; } v; v.f = f;
    unsigned r = v.u + 0x7FFFu + ((v.u >> 16) & 1u);   // round-to-nearest-even
    unsigned short hs = (unsigned short)(r >> 16);
    __bf16 out;
    __builtin_memcpy(&out, &hs, 2);
    return out;
}

static __device__ inline float sigmoidf_(float x) {
    return 1.0f / (1.0f + __expf(-x));
}

// Build a v16bf WMMA fragment for this lane.
// 16-bit A (16x32) / B (32x16) lane layout (CDNA5 ISA 7.12.2):
//   lanes 0-15 : row/col = lane,    halves = K{k0..k0+7},  K{k0+16..k0+23}
//   lanes 16-31: row/col = lane-16, halves = K{k0+8..+15}, K{k0+24..+31}
// Caller passes p = row_base + k0 + khalf  (khalf = (lane>>4)*8).
static __device__ inline v16bf load_frag(const __bf16* p) {
    v8bf lo = *reinterpret_cast<const v8bf*>(p);
    v8bf hi = *reinterpret_cast<const v8bf*>(p + 16);
    v16bf r;
#pragma unroll
    for (int i = 0; i < 8; ++i) { r[i] = lo[i]; r[8 + i] = hi[i]; }
    return r;
}

static __device__ inline v8f wmma_bf16(v16bf a, v16bf b, v8f c) {
    return __builtin_amdgcn_wmma_f32_16x16x32_bf16(
        false, a, false, b, (short)0, c, false, false);
}

// ---------------------------------------------------------------------------
// Kernel 1: weight prep (once per launch).
// Wt[j][k] (bf16, 4096 x 3072, row j contiguous in k) =
//   k <1024 : Wx[k][j] ; k<2048 : Wh[k-1024][j] ; else Wattn[k-2048][j]
// ---------------------------------------------------------------------------
__global__ void prep_weights(const float* __restrict__ Wx,
                             const float* __restrict__ Wh,
                             const float* __restrict__ Wattn,
                             __bf16* __restrict__ Wt) {
    int idx = blockIdx.x * 256 + threadIdx.x;          // 4096*3072 threads
    int j = idx / K_;
    int k = idx - j * K_;
    float v;
    if (k < D_)            v = Wx[k * FH_ + j];
    else if (k < D_ + H_)  v = Wh[(k - D_) * FH_ + j];
    else                   v = Wattn[(k - D_ - H_) * FH_ + j];
    Wt[idx] = f2bf(v);
}

// ---------------------------------------------------------------------------
// Kernel 2: h0 = c0 = mean(A, axis=(2,3))
// ---------------------------------------------------------------------------
__global__ void init_state(const float* __restrict__ A,
                           float* __restrict__ h, float* __restrict__ c) {
    int idx = blockIdx.x * 256 + threadIdx.x;          // N*H threads
    const float* a16 = A + (size_t)idx * 16;
    float s = 0.f;
#pragma unroll
    for (int l = 0; l < 16; ++l) s += a16[l];
    s *= (1.0f / 16.0f);
    h[idx] = s;
    c[idx] = s;
}

// ---------------------------------------------------------------------------
// Kernel 3 (per step): attention + assemble bf16 A-row u[n] = [xt | h | attn]
// one block (256 threads) per sample n
// ---------------------------------------------------------------------------
__global__ void attn_assemble(const float* __restrict__ x,
                              const float* __restrict__ A,
                              const float* __restrict__ hstate,
                              __bf16* __restrict__ u, int t) {
    __shared__ float red[16 * 256];
    __shared__ float wsm[16];
    const int n   = blockIdx.x;
    const int tid = threadIdx.x;

    // scores[l] = sum_h h[n,h] * A[n,h,l]
    float part[16];
#pragma unroll
    for (int l = 0; l < 16; ++l) part[l] = 0.f;
    for (int h = tid; h < H_; h += 256) {
        float hv = hstate[n * H_ + h];
        const float* a16 = A + ((size_t)n * H_ + h) * 16;
#pragma unroll
        for (int l = 0; l < 16; ++l) part[l] += hv * a16[l];
    }
#pragma unroll
    for (int l = 0; l < 16; ++l) red[l * 256 + tid] = part[l];
    __syncthreads();

    if (tid < 16) {                      // reduce + scale
        float s = 0.f;
        for (int i = 0; i < 256; ++i) s += red[tid * 256 + i];
        wsm[tid] = s * 0.03125f;         // 1/sqrt(1024)
    }
    __syncthreads();
    if (tid == 0) {                      // softmax over 16
        float mx = wsm[0];
#pragma unroll
        for (int l = 1; l < 16; ++l) mx = fmaxf(mx, wsm[l]);
        float ssum = 0.f;
        float e[16];
#pragma unroll
        for (int l = 0; l < 16; ++l) { e[l] = __expf(wsm[l] - mx); ssum += e[l]; }
        float inv = 1.0f / ssum;
#pragma unroll
        for (int l = 0; l < 16; ++l) wsm[l] = e[l] * inv;
    }
    __syncthreads();

    float w[16];
#pragma unroll
    for (int l = 0; l < 16; ++l) w[l] = wsm[l];

    __bf16* urow = u + (size_t)n * K_;
    // attn[h] = sum_l A[n,h,l]*w[l] ; also stash h in bf16
    for (int h = tid; h < H_; h += 256) {
        const float* a16 = A + ((size_t)n * H_ + h) * 16;
        float av = 0.f;
#pragma unroll
        for (int l = 0; l < 16; ++l) av += a16[l] * w[l];
        urow[D_ + h]      = f2bf(hstate[n * H_ + h]);
        urow[D_ + H_ + h] = f2bf(av);
    }
    // xt
    const float* xt = x + ((size_t)n * T_ + t) * D_;
    for (int d = tid; d < D_; d += 256) urow[d] = f2bf(xt[d]);
}

// ---------------------------------------------------------------------------
// Kernel 4 (per step): fused gate GEMM (WMMA bf16) + LSTM cell update.
// Each wave owns a 32(M) x 16(gate-N) tile: two m-tiles x 4 gates = 8
// accumulators.  K = 3072 in chunks of 32, software double-buffered so the
// next chunk's 6 fragments (2 A + 4 B) are in flight while 8 WMMAs issue.
// __launch_bounds__(256, 1): trade occupancy for registers (~220 VGPRs live;
// default budget caused scratch spills of the fragment buffers).
// Grid: 64 blocks x 256 thr (8 waves) = 512 wave-tiles.
// ---------------------------------------------------------------------------
__global__ void __launch_bounds__(256, 1)
lstm_step(const __bf16* __restrict__ u,
          const __bf16* __restrict__ Wt,
          const float*  __restrict__ bias,
          float* __restrict__ hstate,
          float* __restrict__ cstate,
          float* __restrict__ out, int t) {
    const int wave   = threadIdx.x >> 5;
    const int lane   = threadIdx.x & 31;
    const int tile   = blockIdx.x * 8 + wave;   // 0..511
    const int tile_m = tile >> 6;               // 0..7  (32 rows each)
    const int tile_n = tile & 63;               // 0..63 (16 gate cols each)
    const int m_base = tile_m * 32;
    const int n_base = tile_n * 16;

    const int colN  = lane & 15;
    const int khalf = (lane >> 4) << 3;         // 0 or 8
    const int jg    = n_base + colN;            // gate column in [0,H)

    const __bf16* arow0 = u + (size_t)(m_base + colN) * K_ + khalf;
    const __bf16* arow1 = arow0 + (size_t)16 * K_;           // second m-tile
    const __bf16* bwI   = Wt + (size_t)(0 * H_ + jg) * K_ + khalf;
    const __bf16* bwF   = Wt + (size_t)(1 * H_ + jg) * K_ + khalf;
    const __bf16* bwO   = Wt + (size_t)(2 * H_ + jg) * K_ + khalf;
    const __bf16* bwG   = Wt + (size_t)(3 * H_ + jg) * K_ + khalf;

    v8f aI0 = {}, aF0 = {}, aO0 = {}, aG0 = {};
    v8f aI1 = {}, aF1 = {}, aO1 = {}, aG1 = {};

    // prologue: fragments for k0 = 0
    v16bf fa0 = load_frag(arow0);
    v16bf fa1 = load_frag(arow1);
    v16bf fbI = load_frag(bwI);
    v16bf fbF = load_frag(bwF);
    v16bf fbO = load_frag(bwO);
    v16bf fbG = load_frag(bwG);

    for (int k0 = 0; k0 < K_ - 32; k0 += 32) {
        const int kn = k0 + 32;
        // prefetch ahead (-> global_prefetch_b8, keeps weight stream hot)
        __builtin_prefetch(bwI + k0 + 512, 0, 3);
        __builtin_prefetch(bwF + k0 + 512, 0, 3);
        __builtin_prefetch(bwO + k0 + 512, 0, 3);
        __builtin_prefetch(bwG + k0 + 512, 0, 3);

        // issue next chunk's loads before consuming current fragments
        v16bf na0 = load_frag(arow0 + kn);
        v16bf na1 = load_frag(arow1 + kn);
        v16bf nbI = load_frag(bwI + kn);
        v16bf nbF = load_frag(bwF + kn);
        v16bf nbO = load_frag(bwO + kn);
        v16bf nbG = load_frag(bwG + kn);

        // 8 WMMAs on current chunk (A-fragments reused x4, B-fragments x2)
        aI0 = wmma_bf16(fa0, fbI, aI0);
        aI1 = wmma_bf16(fa1, fbI, aI1);
        aF0 = wmma_bf16(fa0, fbF, aF0);
        aF1 = wmma_bf16(fa1, fbF, aF1);
        aO0 = wmma_bf16(fa0, fbO, aO0);
        aO1 = wmma_bf16(fa1, fbO, aO1);
        aG0 = wmma_bf16(fa0, fbG, aG0);
        aG1 = wmma_bf16(fa1, fbG, aG1);

        fa0 = na0; fa1 = na1;
        fbI = nbI; fbF = nbF; fbO = nbO; fbG = nbG;
    }
    // drain: last chunk
    aI0 = wmma_bf16(fa0, fbI, aI0);
    aI1 = wmma_bf16(fa1, fbI, aI1);
    aF0 = wmma_bf16(fa0, fbF, aF0);
    aF1 = wmma_bf16(fa1, fbF, aF1);
    aO0 = wmma_bf16(fa0, fbO, aO0);
    aO1 = wmma_bf16(fa1, fbO, aO1);
    aG0 = wmma_bf16(fa0, fbG, aG0);
    aG1 = wmma_bf16(fa1, fbG, aG1);

    // C/D layout: element r <-> M = m_tilebase + 8*(lane>=16) + r, N = colN.
    const int moff = (lane >> 4) << 3;
    const float bI = bias[0 * H_ + jg];
    const float bF = bias[1 * H_ + jg];
    const float bO = bias[2 * H_ + jg];
    const float bG = bias[3 * H_ + jg];

#pragma unroll
    for (int half = 0; half < 2; ++half) {
        const int mb = m_base + half * 16 + moff;
        const v8f& AI = half ? aI1 : aI0;
        const v8f& AF = half ? aF1 : aF0;
        const v8f& AO = half ? aO1 : aO0;
        const v8f& AG = half ? aG1 : aG0;
#pragma unroll
        for (int r = 0; r < 8; ++r) {
            const int m = mb + r;
            float vi = sigmoidf_(AI[r] + bI);
            float vf = sigmoidf_(AF[r] + bF);
            float vo = sigmoidf_(AO[r] + bO);
            float vg = tanhf(AG[r] + bG);
            float cp = cstate[m * H_ + jg];
            float cn = vf * cp + vi * vg;
            float hn = vo * tanhf(cn);
            cstate[m * H_ + jg] = cn;
            hstate[m * H_ + jg] = hn;
            out[((size_t)m * T_ + t) * H_ + jg] = hn;
        }
    }
}

// ---------------------------------------------------------------------------
// Host launcher
// ---------------------------------------------------------------------------
extern "C" void kernel_launch(void* const* d_in, const int* in_sizes, int n_in,
                              void* d_out, int out_size, void* d_ws, size_t ws_size,
                              hipStream_t stream) {
    const float* x     = (const float*)d_in[0];
    const float* A     = (const float*)d_in[1];
    const float* Wx    = (const float*)d_in[2];
    const float* Wh    = (const float*)d_in[3];
    const float* Wattn = (const float*)d_in[4];
    const float* bias  = (const float*)d_in[5];
    float* out = (float*)d_out;

    // workspace layout (all offsets 1KB aligned): 24MB + 1MB + 1MB + 1.5MB
    char* wsb = (char*)d_ws;
    __bf16* Wt     = (__bf16*)(wsb);                          // 4096*3072 bf16
    float*  hstate = (float*) (wsb + 25165824);               // N*H f32
    float*  cstate = (float*) (wsb + 26214400);               // N*H f32
    __bf16* u      = (__bf16*)(wsb + 27262976);               // N*K bf16

    prep_weights<<<(FH_ * K_) / 256, 256, 0, stream>>>(Wx, Wh, Wattn, Wt);
    init_state<<<(N_ * H_) / 256, 256, 0, stream>>>(A, hstate, cstate);

    for (int t = 0; t < T_; ++t) {
        attn_assemble<<<N_, 256, 0, stream>>>(x, A, hstate, u, t);
        lstm_step<<<64, 256, 0, stream>>>(u, Wt, bias, hstate, cstate, out, t);
    }
}
